// Convnet_86998857548113
// MI455X (gfx1250) — matrix operand order repaired and checked by reading, not compiled, not run on hardware
//
#include <hip/hip_runtime.h>

typedef __attribute__((ext_vector_type(16))) _Float16 v16h;
typedef __attribute__((ext_vector_type(8)))  _Float16 v8h;
typedef __attribute__((ext_vector_type(4)))  _Float16 v4h;
typedef __attribute__((ext_vector_type(8)))  float    v8f;

#define N_SECTIONS 40
#define L_POST     20
#define NFM        100
#define FREQ       40
#define NT_        1000
#define T_STEPS    32
#define L_PRE      25                    // KT + L_POST - 1
#define KDIM       48                    // KT*KF
#define KPAD       64                    // padded K (2 WMMA k-steps)
#define FPAD       112                   // padded features (7 tiles of 16)
#define XROWS      28                    // 25 real rows + 3 zero pad rows
#define POOL_W     4
#define UNITS_PER_SEC (NFM * POOL_W)     // 400
#define TOTAL_UNITS   (N_SECTIONS * UNITS_PER_SEC) // 16000
#define NTILES_F   7                     // feature tiles (112/16)
#define NTILES_P   40                    // spatial tiles (640/16)

// One workgroup per (section s, timestep t); 256 threads = 8 wave32 waves.
// GEMM D[640 spatial x 112 feat] = im2col(X)[640x48] x W_s^T[48x112],
// f16 in / f32 accum via v_wmma_f32_16x16x32_f16. Spatial index is
// window-major (p = w*160 + y*8 + xi, x = 8w+xi) so each lane's 8 D rows sit
// in one pool-window row: fused fire(>=15)+maxpool = fmax tree + 1 compare
// per tile, 1 ds_or per p-tile. All 14 WMMAs per p-tile issue into 7
// independent accumulators before any D is read, so the WMMA->VALU hazard
// slots are filled with real work instead of v_nops.
__global__ __launch_bounds__(256) void snn_conv_pool_kernel(
    const float* __restrict__ X,   // [T, 1, NT, FREQ] f32
    const float* __restrict__ W,   // [S, NFM, 1, KT, KF] f32
    float* __restrict__ out)       // [1 + S*T*NFM*4] f32
{
  const int t     = blockIdx.x;
  const int s     = blockIdx.y;
  const int tid   = threadIdx.x;
  const int lane  = tid & 31;
  const int wave  = tid >> 5;           // 0..7
  const int lhalf = lane >> 4;          // 0 or 1
  const int lmod  = lane & 15;

  __shared__ __align__(16) _Float16 Ws[FPAD * KPAD];   // 14336 B, zero padded
  __shared__ __align__(16) _Float16 Xs[XROWS * FREQ];  //  2240 B, zero padded
  __shared__ unsigned int Pool2[POOL_W * 16];          // [w][lane] f-tile bits

  // ---- zero-init padded LDS (all padding finite/zero) ----
  unsigned int* wz = (unsigned int*)Ws;
  for (int i = tid; i < (FPAD * KPAD) / 2; i += 256) wz[i] = 0u;
  unsigned int* xz = (unsigned int*)Xs;
  for (int i = tid; i < (XROWS * FREQ) / 2; i += 256) xz[i] = 0u;
  if (tid < POOL_W * 16) Pool2[tid] = 0u;
  __syncthreads();

  // ---- stage: W section (f32->f16, [100][48] -> padded [112][64]) ----
  const float4* Wg4 = (const float4*)(W + (size_t)s * (NFM * KDIM));
  for (int i = tid; i < (NFM * KDIM) / 4; i += 256) {
    float4 v = Wg4[i];
    const int flat = i * 4;
    const int m = flat / KDIM;
    const int k = flat - m * KDIM;       // multiple of 4 -> stays in-row
    v4h h = { (_Float16)v.x, (_Float16)v.y, (_Float16)v.z, (_Float16)v.w };
    *(v4h*)(Ws + m * KPAD + k) = h;      // 8B-aligned ds_store_b64
  }
  // ---- stage: X slice (1000 contiguous f32 -> f16, rows 25..27 stay 0) ----
  const float4* Xg4 = (const float4*)(X + (size_t)t * (NT_ * FREQ) +
                                      (size_t)s * (L_PRE * FREQ));
  for (int i = tid; i < (L_PRE * FREQ) / 4; i += 256) {
    float4 v = Xg4[i];
    v4h h = { (_Float16)v.x, (_Float16)v.y, (_Float16)v.z, (_Float16)v.w };
    *(v4h*)(Xs + i * 4) = h;
  }
  __syncthreads();

  // ---- hoist B (weight) fragments: 7 f-tiles x 2 k-steps ----
  // B 32x16 f16 layout: lane = column f = lane&15; k(e) = (lane>>4)*8 + (e&7) + ((e>>3)<<4)
  // -> two 8-k runs per lane = two aligned ds_load_b128 from Ws row.
  v16h bfrag[NTILES_F * 2];
  #pragma unroll
  for (int ft = 0; ft < NTILES_F; ++ft) {
    #pragma unroll
    for (int ks = 0; ks < 2; ++ks) {
      const _Float16* rp = Ws + (ft * 16 + lmod) * KPAD + ks * 32 + lhalf * 8;
      v8h lo = *(const v8h*)(rp);
      v8h hi = *(const v8h*)(rp + 16);
      v16h b;
      #pragma unroll
      for (int e = 0; e < 8; ++e) { b[e] = lo[e]; b[8 + e] = hi[e]; }
      bfrag[ft * 2 + ks] = b;
    }
  }

  // ---- main loop: 40 p-tiles, exactly 5 per wave (wave-uniform) ----
  for (int pt = wave; pt < NTILES_P; pt += 8) {
    const int w   = pt / 10;                 // pool window, uniform per tile
    const int p   = pt * 16 + lmod;          // spatial index
    const int rem = p - w * 160;             // y*8 + xi
    const int y   = rem >> 3;
    const int x   = (w << 3) + (rem & 7);    // <= 31

    // A 16x32 f16 layout: lane = row p = lane&15;
    //   k(e) = ks*32 + (lane>>4)*8 + (e&7) + ((e>>3)<<4)
    // -> rows (y + ks*4 + lhalf) and (+2) of Xs, cols x..x+7 (two 8-f16 runs).
    v16h afr[2];
    #pragma unroll
    for (int ks = 0; ks < 2; ++ks) {
      const _Float16* r0 = Xs + (y + ks * 4 + lhalf) * FREQ + x;
      v16h a;
      #pragma unroll
      for (int e = 0; e < 8; ++e) { a[e] = r0[e]; a[8 + e] = r0[2 * FREQ + e]; }
      afr[ks] = a;
    }

    // Phase 1: issue all 14 WMMAs into 7 independent accumulators.
    v8f accs[NTILES_F];
    #pragma unroll
    for (int ft = 0; ft < NTILES_F; ++ft) {
      v8f acc = {};
      acc = __builtin_amdgcn_wmma_f32_16x16x32_f16(
          false, afr[0], false, bfrag[ft * 2 + 0], (short)0, acc, false, false);
      acc = __builtin_amdgcn_wmma_f32_16x16x32_f16(
          false, afr[1], false, bfrag[ft * 2 + 1], (short)0, acc, false, false);
      accs[ft] = acc;
    }

    // Phase 2: per tile, lane's 8 D rows = 8 x-positions of one pool-window
    // row: max-reduce, one threshold compare, set bit ft (f = ft*16 + lmod).
    unsigned int fbits = 0u;
    #pragma unroll
    for (int ft = 0; ft < NTILES_F; ++ft) {
      const v8f acc = accs[ft];
      float m01 = fmaxf(acc[0], acc[1]);
      float m23 = fmaxf(acc[2], acc[3]);
      float m45 = fmaxf(acc[4], acc[5]);
      float m67 = fmaxf(acc[6], acc[7]);
      float mx  = fmaxf(fmaxf(m01, m23), fmaxf(m45, m67));
      fbits |= (mx >= 15.0f) ? (1u << ft) : 0u;
    }
    atomicOr(&Pool2[w * 16 + lmod], fbits);   // ds_or_b32, once per p-tile
  }
  __syncthreads();

  // ---- expand pool bits to 0/1 floats: pool[s,t,f,0,w] ----
  float* Og = out + 1 + ((size_t)s * T_STEPS + t) * (size_t)UNITS_PER_SEC;
  for (int i = tid; i < UNITS_PER_SEC; i += 256) {
    const int f = i >> 2, w = i & 3;
    Og[i] = (float)((Pool2[w * 16 + (f & 15)] >> (f >> 4)) & 1u);
  }
}

// Winner: min over units of key = first_spike_t*16000 + flat_index; matches
// jnp.argmax (earliest spike, ties -> lowest flat index). Writes section or -1.
__global__ __launch_bounds__(256) void snn_winner_kernel(float* __restrict__ out) {
  __shared__ int smin[256];
  const int tid = threadIdx.x;
  int best = 0x7fffffff;
  for (int unit = tid; unit < TOTAL_UNITS; unit += 256) {
    const int s = unit / UNITS_PER_SEC;
    const int u = unit - s * UNITS_PER_SEC;
    const float* p = out + 1 + (size_t)s * (T_STEPS * UNITS_PER_SEC) + u;
    int ft = -1;
    for (int tt = 0; tt < T_STEPS; ++tt) {
      if (p[(size_t)tt * UNITS_PER_SEC] > 0.0f) { ft = tt; break; }
    }
    if (ft >= 0) {
      const int key = ft * TOTAL_UNITS + unit;
      best = best < key ? best : key;
    }
  }
  smin[tid] = best;
  __syncthreads();
  for (int off = 128; off > 0; off >>= 1) {
    if (tid < off) {
      const int o = smin[tid + off];
      if (o < smin[tid]) smin[tid] = o;
    }
    __syncthreads();
  }
  if (tid == 0) {
    const int k = smin[0];
    out[0] = (k == 0x7fffffff) ? -1.0f
                               : (float)((k % TOTAL_UNITS) / UNITS_PER_SEC);
  }
}

extern "C" void kernel_launch(void* const* d_in, const int* in_sizes, int n_in,
                              void* d_out, int out_size, void* d_ws, size_t ws_size,
                              hipStream_t stream) {
  const float* X = (const float*)d_in[0];   // [32, 1, 1000, 40] f32
  const float* W = (const float*)d_in[1];   // [40, 100, 1, 6, 8] f32
  float* out = (float*)d_out;               // 1 + 512000 floats

  dim3 grid(T_STEPS, N_SECTIONS);           // 1280 workgroups
  snn_conv_pool_kernel<<<grid, 256, 0, stream>>>(X, W, out);
  snn_winner_kernel<<<1, 256, 0, stream>>>(out);
}
